// MambaVision_36326833390111
// MI455X (gfx1250) — matrix operand order repaired and checked by reading, not compiled, not run on hardware
//
#include <hip/hip_runtime.h>
#include <hip/hip_bf16.h>

// ---------------------------------------------------------------------------
// Mamba block for MI455X (gfx1250, wave32, WMMA).
// Big GEMMs -> v_wmma_f32_16x16x32_bf16 with LDS-staged 128x64 tiles.
// Scan / conv / softplus -> f32 VALU.
// ---------------------------------------------------------------------------

#define BATCH   16
#define SEQLEN  1024
#define D_MODEL 512
#define D_STATE 16
#define D_CONV  4
#define D_INNER 1024
#define D_HALF  512
#define DT_RANK 32
#define MROWS   (BATCH * SEQLEN)   // 16384

typedef __attribute__((ext_vector_type(16))) __bf16          v16bf;
typedef __attribute__((ext_vector_type(8)))  float           v8f;
typedef __attribute__((ext_vector_type(8)))  unsigned short  us8;

union frag_u { v16bf v; us8 h[2]; };

__device__ __forceinline__ unsigned short f32_to_bf16(float f) {
    unsigned u = __float_as_uint(f);
    unsigned r = u + 0x7FFFu + ((u >> 16) & 1u);   // round-to-nearest-even
    return (unsigned short)(r >> 16);
}

// ---------------------------------------------------------------------------
// f32 -> bf16 bulk conversion
// ---------------------------------------------------------------------------
__global__ __launch_bounds__(256) void cvt_f32_bf16(const float* __restrict__ in,
                                                    unsigned short* __restrict__ out,
                                                    int n) {
    int i = blockIdx.x * 256 + threadIdx.x;
    if (i < n) out[i] = f32_to_bf16(in[i]);
}

// ---------------------------------------------------------------------------
// Generic bf16 WMMA GEMM:  C[M][N](f32) = A[M][K](bf16) * B^T, with B given as
// Bt[N][K] bf16 row-major (exactly the layout of the torch-style weights).
// Workgroup: 256 threads = 8 waves; tile 128(M) x 64(N); K-step 32.
// Wave w computes rows [16w,16w+16) x all 64 cols (4 wmma accumulators).
// Requires M%128==0, N%64==0, K%32==0 (true for all three GEMMs here).
// ---------------------------------------------------------------------------
#define TM 128
#define TN 64
#define TK 32

__global__ __launch_bounds__(256) void gemm_bf16_wmma(
    const unsigned short* __restrict__ A,  int lda,
    const unsigned short* __restrict__ Bt, int ldb,
    float* __restrict__ C, int ldc,
    int M, int N, int K)
{
    __shared__ unsigned short lA[TM * TK];   // 8 KB
    __shared__ unsigned short lB[TN * TK];   // 4 KB

    const int tid  = threadIdx.x;
    const int wave = tid >> 5;
    const int lane = tid & 31;
    const int m0   = blockIdx.y * TM;
    const int n0   = blockIdx.x * TN;

    v8f acc[4] = { v8f{0.f}, v8f{0.f}, v8f{0.f}, v8f{0.f} };

    const int arow = tid >> 1;                 // 0..127
    const int acol = (tid & 1) * 16;           // 0 / 16
    const int brow = tid >> 2;                 // 0..63
    const int bcol = (tid & 3) * 8;            // 0/8/16/24

    for (int k0 = 0; k0 < K; k0 += TK) {
        // --- stage A tile 128x32 (each thread: 16 contiguous bf16 = 32B) ---
        {
            const unsigned short* src = A + (size_t)(m0 + arow) * lda + k0 + acol;
            *(us8*)&lA[arow * TK + acol]     = *(const us8*)(src);
            *(us8*)&lA[arow * TK + acol + 8] = *(const us8*)(src + 8);
        }
        // --- stage B tile [n][k] 64x32 (each thread: 8 contiguous bf16) ---
        {
            const unsigned short* src = Bt + (size_t)(n0 + brow) * ldb + k0 + bcol;
            *(us8*)&lB[brow * TK + bcol] = *(const us8*)(src);
        }
        if (k0 + TK < K) {  // hint next tiles toward the caches
            __builtin_prefetch(A  + (size_t)(m0 + arow) * lda + k0 + TK + acol, 0, 1);
            __builtin_prefetch(Bt + (size_t)(n0 + brow) * ldb + k0 + TK + bcol, 0, 1);
        }
        __syncthreads();

        // --- A fragment (ISA 7.12.2 bf16 16x32 layout) ---
        frag_u fa;
        {
            int m  = wave * 16 + (lane & 15);
            int kb = (lane >> 4) * 8;          // 0 or 8
            fa.h[0] = *(const us8*)&lA[m * TK + kb];        // K = kb..kb+7
            fa.h[1] = *(const us8*)&lA[m * TK + kb + 16];   // K = kb+16..kb+23
        }
        // --- 4 WMMAs reusing the A fragment ---
        #pragma unroll
        for (int c = 0; c < 4; ++c) {
            frag_u fb;
            int n  = c * 16 + (lane & 15);
            int kb = (lane >> 4) * 8;          // 0 or 8
            fb.h[0] = *(const us8*)&lB[n * TK + kb];        // K = kb..kb+7
            fb.h[1] = *(const us8*)&lB[n * TK + kb + 16];   // K = kb+16..kb+23
            acc[c] = __builtin_amdgcn_wmma_f32_16x16x32_bf16(
                         false, fa.v, false, fb.v, (short)0, acc[c], false, false);
        }
        __syncthreads();
    }

    // --- store (C/D layout: VGPR r -> row r + 8*(lane>>4), col = lane&15) ---
    const int ncol  = lane & 15;
    const int mbase = m0 + wave * 16 + (lane >> 4) * 8;
    #pragma unroll
    for (int c = 0; c < 4; ++c) {
        const int col = n0 + c * 16 + ncol;
        #pragma unroll
        for (int r = 0; r < 8; ++r)
            C[(size_t)(mbase + r) * ldc + col] = acc[c][r];
    }
}

// ---------------------------------------------------------------------------
// Depthwise conv1d (K=4, 'same': taps l-1..l+2) + SiLU for both x and z halves
// of xz[M][1024]. Emits x as f32 (for the scan) + bf16 (for x_proj GEMM) and
// z as bf16 into the concat buffer yz[M][1024] second half.
// ---------------------------------------------------------------------------
__global__ __launch_bounds__(256) void conv_silu_kernel(
    const float* __restrict__ xz,
    const float* __restrict__ wx,    // [512][4]
    const float* __restrict__ wz,    // [512][4]
    float* __restrict__ xc_f32,
    unsigned short* __restrict__ xc_bf16,
    unsigned short* __restrict__ yz_bf16)
{
    int idx = blockIdx.x * 256 + threadIdx.x;      // B*L*512
    int c = idx & (D_HALF - 1);
    int l = (idx >> 9) & (SEQLEN - 1);
    int b = idx >> 19;
    const int row = b * SEQLEN + l;

    float ax = 0.f, az = 0.f;
    #pragma unroll
    for (int t = 0; t < D_CONV; ++t) {
        int ls = l + t - 1;
        if (ls >= 0 && ls < SEQLEN) {
            const float* p = xz + (size_t)(b * SEQLEN + ls) * D_INNER;
            ax += p[c]          * wx[c * D_CONV + t];
            az += p[c + D_HALF] * wz[c * D_CONV + t];
        }
    }
    float sx = ax / (1.f + __expf(-ax));
    float sz = az / (1.f + __expf(-az));
    xc_f32 [(size_t)row * D_HALF + c] = sx;
    xc_bf16[(size_t)row * D_HALF + c] = f32_to_bf16(sx);
    yz_bf16[(size_t)row * D_INNER + D_HALF + c] = f32_to_bf16(sz);
}

// ---------------------------------------------------------------------------
// delta = softplus(dt @ dt_proj_w^T + b); dt = x_dbl[:, :32]. One block per
// token row (512 threads = 512 channels), dt staged in LDS.
// ---------------------------------------------------------------------------
__global__ __launch_bounds__(512) void delta_kernel(
    const float* __restrict__ x_dbl,   // [M][64]
    const float* __restrict__ dtw,     // [512][32]
    const float* __restrict__ dtb,     // [512]
    float* __restrict__ delta)         // [M][512]
{
    __shared__ float sdt[DT_RANK];
    const int m = blockIdx.x;
    const int d = threadIdx.x;
    if (d < DT_RANK) sdt[d] = x_dbl[(size_t)m * 64 + d];
    __syncthreads();

    const float* w = dtw + d * DT_RANK;
    float a = dtb[d];
    #pragma unroll
    for (int r = 0; r < DT_RANK; ++r) a += sdt[r] * w[r];
    float sp = (a > 20.f) ? a : log1pf(__expf(a));
    delta[(size_t)m * D_HALF + d] = sp;
}

// ---------------------------------------------------------------------------
// Selective scan. One thread per (b,d) chain: h[16] in registers; B/C vectors
// (shared across all d of a batch) staged through LDS in 16-step chunks.
// y (+ D*x skip) written as bf16 into first half of concat buffer yz[M][1024].
// ---------------------------------------------------------------------------
__global__ __launch_bounds__(128) void scan_kernel(
    const float* __restrict__ xc,      // [M][512]
    const float* __restrict__ delta,   // [M][512]
    const float* __restrict__ x_dbl,   // [M][64]  (B: cols 32..47, C: 48..63)
    const float* __restrict__ A_log,   // [512][16]
    const float* __restrict__ Dvec,    // [512]
    unsigned short* __restrict__ yz_bf16)
{
    const int b = blockIdx.x;                       // 0..15
    const int d = blockIdx.y * 128 + threadIdx.x;   // 0..511

    __shared__ float sB[16 * D_STATE];
    __shared__ float sC[16 * D_STATE];

    float a[D_STATE], h[D_STATE];
    #pragma unroll
    for (int n = 0; n < D_STATE; ++n) {
        a[n] = -__expf(A_log[d * D_STATE + n]);
        h[n] = 0.f;
    }
    const float dval = Dvec[d];

    for (int l0 = 0; l0 < SEQLEN; l0 += 16) {
        for (int i = threadIdx.x; i < 16 * D_STATE; i += 128) {
            int step = i >> 4, n = i & 15;
            size_t m = (size_t)(b * SEQLEN + l0 + step);
            sB[i] = x_dbl[m * 64 + 32 + n];
            sC[i] = x_dbl[m * 64 + 48 + n];
        }
        __syncthreads();

        for (int s = 0; s < 16; ++s) {
            const size_t m  = (size_t)(b * SEQLEN + l0 + s);
            const float dlt = delta[m * D_HALF + d];
            const float xv  = xc   [m * D_HALF + d];
            float y = 0.f;
            #pragma unroll
            for (int n = 0; n < D_STATE; ++n) {
                float dA = __expf(dlt * a[n]);
                h[n] = dA * h[n] + dlt * sB[s * D_STATE + n] * xv;
                y += h[n] * sC[s * D_STATE + n];
            }
            y += xv * dval;
            yz_bf16[m * D_INNER + d] = f32_to_bf16(y);
        }
        __syncthreads();
    }
}

// ---------------------------------------------------------------------------
// Host-side orchestration
// ---------------------------------------------------------------------------
extern "C" void kernel_launch(void* const* d_in, const int* in_sizes, int n_in,
                              void* d_out, int out_size, void* d_ws, size_t ws_size,
                              hipStream_t stream) {
    const float* hs    = (const float*)d_in[0];   // [16,1024,512]
    const float* w_in  = (const float*)d_in[1];   // [1024,512]
    const float* w_cx  = (const float*)d_in[2];   // [512,1,4]
    const float* w_cz  = (const float*)d_in[3];   // [512,1,4]
    const float* w_xp  = (const float*)d_in[4];   // [64,512]
    const float* w_dt  = (const float*)d_in[5];   // [512,32]
    const float* b_dt  = (const float*)d_in[6];   // [512]
    const float* A_log = (const float*)d_in[7];   // [512,16]
    const float* Dvec  = (const float*)d_in[8];   // [512]
    const float* w_out = (const float*)d_in[9];   // [512,1024]
    float* out = (float*)d_out;                   // [16,1024,512] f32

    char* ws = (char*)d_ws;
    size_t off = 0;
    auto carve = [&](size_t bytes) -> char* {
        char* p = ws + off;
        off += (bytes + 255) & ~(size_t)255;
        return p;
    };

    unsigned short* hs_bf   = (unsigned short*)carve((size_t)MROWS * D_MODEL * 2);
    unsigned short* win_bf  = (unsigned short*)carve((size_t)D_INNER * D_MODEL * 2);
    unsigned short* wxp_bf  = (unsigned short*)carve((size_t)64 * D_HALF * 2);
    unsigned short* wout_bf = (unsigned short*)carve((size_t)D_MODEL * D_INNER * 2);
    float*          xz      = (float*)         carve((size_t)MROWS * D_INNER * 4);
    float*          xc      = (float*)         carve((size_t)MROWS * D_HALF * 4);
    unsigned short* xc_bf   = (unsigned short*)carve((size_t)MROWS * D_HALF * 2);
    float*          x_dbl   = (float*)         carve((size_t)MROWS * 64 * 4);
    float*          delta   = (float*)         carve((size_t)MROWS * D_HALF * 4);
    unsigned short* yz_bf   = (unsigned short*)carve((size_t)MROWS * D_INNER * 2);

    // 0) weight / activation bf16 conversion
    {
        int n;
        n = MROWS * D_MODEL;
        cvt_f32_bf16<<<(n + 255) / 256, 256, 0, stream>>>(hs, hs_bf, n);
        n = D_INNER * D_MODEL;
        cvt_f32_bf16<<<(n + 255) / 256, 256, 0, stream>>>(w_in, win_bf, n);
        n = 64 * D_HALF;
        cvt_f32_bf16<<<(n + 255) / 256, 256, 0, stream>>>(w_xp, wxp_bf, n);
        n = D_MODEL * D_INNER;
        cvt_f32_bf16<<<(n + 255) / 256, 256, 0, stream>>>(w_out, wout_bf, n);
    }

    // 1) xz = hs @ in_proj_w^T : M=16384 N=1024 K=512
    gemm_bf16_wmma<<<dim3(D_INNER / TN, MROWS / TM), 256, 0, stream>>>(
        hs_bf, D_MODEL, win_bf, D_MODEL, xz, D_INNER, MROWS, D_INNER, D_MODEL);

    // 2) depthwise conv + SiLU (x -> f32+bf16, z -> bf16 into concat buffer)
    {
        int n = MROWS * D_HALF;
        conv_silu_kernel<<<(n + 255) / 256, 256, 0, stream>>>(
            xz, w_cx, w_cz, xc, xc_bf, yz_bf);
    }

    // 3) x_dbl = x @ x_proj_w^T : M=16384 N=64 K=512
    gemm_bf16_wmma<<<dim3(64 / TN, MROWS / TM), 256, 0, stream>>>(
        xc_bf, D_HALF, wxp_bf, D_HALF, x_dbl, 64, MROWS, 64, D_HALF);

    // 4) delta = softplus(dt @ dt_proj_w^T + b)
    delta_kernel<<<MROWS, 512, 0, stream>>>(x_dbl, w_dt, b_dt, delta);

    // 5) selective scan -> y (bf16) into concat buffer
    scan_kernel<<<dim3(BATCH, D_HALF / 128), 128, 0, stream>>>(
        xc, delta, x_dbl, A_log, Dvec, yz_bf);

    // 6) out = concat(y,z) @ out_proj_w^T : M=16384 N=512 K=1024 (f32 out)
    gemm_bf16_wmma<<<dim3(D_MODEL / TN, MROWS / TM), 256, 0, stream>>>(
        yz_bf, D_INNER, wout_bf, D_INNER, out, D_MODEL, MROWS, D_MODEL, D_INNER);
}
